// TransformerCNNObjectInference_45878840656289
// MI455X (gfx1250) — matrix operand order repaired and checked by reading, not compiled, not run on hardware
//
#include <hip/hip_runtime.h>
#include <hip/hip_bf16.h>
#include <math.h>

// ---------------------------------------------------------------------------
// Types for WMMA (gfx1250, wave32)
// ---------------------------------------------------------------------------
typedef __attribute__((ext_vector_type(16))) _Float16 h16;
typedef __attribute__((ext_vector_type(8)))  _Float16 h8;
typedef __attribute__((ext_vector_type(4)))  _Float16 h4;
typedef __attribute__((ext_vector_type(8)))  float    v8f;

#define BB   64
#define DD   128
#define KK   10
#define HH   16
#define WW   24
#define HWN  384          // 16*24
#define BKN  640          // B*K

// Async global->LDS copy, 16B per lane (GLOBAL_LOAD_ASYNC_TO_LDS_B128).
// ldsOff: wave-relative LDS byte offset (low 32 bits of flat shared pointer).
__device__ __forceinline__ void async_copy_b128(unsigned ldsOff, const void* gaddr)
{
    asm volatile("global_load_async_to_lds_b128 %0, %1, off"
                 :
                 : "v"(ldsOff), "v"(gaddr)
                 : "memory");
}
__device__ __forceinline__ void wait_asynccnt0()
{
    asm volatile("s_wait_asynccnt 0x0" ::: "memory");
}

// ===========================================================================
// Implicit-GEMM conv using v_wmma_f32_16x16x32_f16.
//   in   : f16 [Nb][Hin][Win][CPin]   (CPin multiple of 32, padded w/ zeros)
//   wpk  : f16 [Mpad][ksz*ksz*CPin]  (rows >= cout are zero)
//   out  : f16 [Nb*Hout*Wout][CPout] (channels cout..CPout-1 come out zero)
//
// Block: 256 threads = 8 waves.  Block tile M=64 x N=128.
// Wave grid 2x4; each wave owns a 32x32 tile = 4 accumulators
// => 4 WMMAs per K-chunk per wave, 2 LDS b128 reads per WMMA.
// LDS staging double-buffered; A (weights) staged via async global->LDS,
// B (predicated im2col gather) staged via vector path. One barrier per chunk.
// ===========================================================================
__global__ __launch_bounds__(256)
void conv_wmma_kernel(const _Float16* __restrict__ in,
                      const _Float16* __restrict__ wpk,
                      const float*    __restrict__ biasP,
                      _Float16*       __restrict__ out,
                      int Nb, int Hin, int Win, int Hout, int Wout,
                      int CPin, int CPout, int ksz, int stride,
                      int padH, int padW, int doRelu)
{
    const int Npos    = Nb * Hout * Wout;
    const int Ktot    = ksz * ksz * CPin;
    const int cpt     = CPin >> 5;            // 32-wide chunks per filter tap
    const int nChunks = ksz * ksz * cpt;

    __shared__ __align__(16) _Float16 As[2][64][40];    // [m][k], +8 pad
    __shared__ __align__(16) _Float16 Bs[2][128][40];   // [n][k], +8 pad

    const int tid  = threadIdx.x;
    const int lane = tid & 31;
    const int wv   = tid >> 5;
    const int mi2  = wv >> 2;                 // 0..1 : M 32-row half
    const int ni2  = wv & 3;                  // 0..3 : N 32-col quarter
    const int hi   = lane >> 4;
    const int l16  = lane & 15;

    const int posBase = blockIdx.x * 128;
    const int mBase   = blockIdx.y * 64;

    // A-stage mapping: 64 rows x 32 halfs, one async 16B copy per thread
    const int arow = tid >> 2;
    const int aseg = (tid & 3) * 8;
    const _Float16* aPtr = wpk + (size_t)(mBase + arow) * Ktot + aseg;
    const unsigned aLds0 = (unsigned)(uintptr_t)&As[0][arow][aseg];
    const unsigned aLds1 = (unsigned)(uintptr_t)&As[1][arow][aseg];

    // B-stage mapping: 128 positions x 32 halfs, two 16B loads per thread
    const int bn  = tid >> 1;
    const int c16 = (tid & 1) * 16;
    const int gpos   = posBase + bn;
    const bool pvalid = gpos < Npos;
    int iyBase = 0, ixBase = 0;
    const _Float16* inNb = in;
    if (pvalid) {
        const int nb = gpos / (Hout * Wout);
        const int r  = gpos - nb * (Hout * Wout);
        const int oy = r / Wout, ox = r - (r / Wout) * Wout;
        iyBase = oy * stride - padH;
        ixBase = ox * stride - padW;
        inNb   = in + (size_t)nb * Hin * Win * CPin + c16;
    }

    v8f acc00 = {}, acc01 = {}, acc10 = {}, acc11 = {};

    // incremental chunk -> (tap, cb) counters for the *staged* chunk
    int tapS = 0, cbS = 0, k0S = 0;

    auto stage = [&](int buf) {
        // weights: async global->LDS, 16B per lane
        async_copy_b128(buf ? aLds1 : aLds0, aPtr + k0S);
        // im2col gather: two contiguous 16B channel reads (zero-filled OOB)
        const int ky = tapS / ksz, kx = tapS - (tapS / ksz) * ksz;
        const int iy = iyBase + ky;
        const int ix = ixBase + kx;
        h8 b0 = {}, b1 = {};
        if (pvalid && iy >= 0 && iy < Hin && ix >= 0 && ix < Win) {
            const _Float16* src = inNb + (size_t)(iy * Win + ix) * CPin + (cbS << 5);
            b0 = *(const h8*)src;
            b1 = *(const h8*)(src + 8);
        }
        *(h8*)&Bs[buf][bn][c16]     = b0;
        *(h8*)&Bs[buf][bn][c16 + 8] = b1;
        // advance staged-chunk counters
        k0S += 32;
        if (++cbS == cpt) { cbS = 0; ++tapS; }
    };

    // fragment loads per CDNA5 WMMA VGPR layouts:
    //   A lane m=lane&15, K runs {hi*8..+7, 16+hi*8..+7}
    //   B lane n=lane&15, K runs {hi*8..+7, 16+hi*8..+7}
    const int mr0 = mi2 * 32 + l16;
    const int mr1 = mr0 + 16;
    const int nr0 = ni2 * 32 + l16;
    const int nr1 = nr0 + 16;

    stage(0);
    for (int q = 0; q < nChunks; ++q) {
        // async A-copies issued for the buffer about to be published must be
        // LDS-visible before the barrier (compiler can't see inline-asm async).
        wait_asynccnt0();
        __syncthreads();
        const int buf = q & 1;
        if (q + 1 < nChunks) stage(buf ^ 1);

        h8 a00 = *(const h8*)&As[buf][mr0][hi * 8];
        h8 a01 = *(const h8*)&As[buf][mr0][16 + hi * 8];
        h8 a10 = *(const h8*)&As[buf][mr1][hi * 8];
        h8 a11 = *(const h8*)&As[buf][mr1][16 + hi * 8];
        h8 b00 = *(const h8*)&Bs[buf][nr0][hi * 8];
        h8 b01 = *(const h8*)&Bs[buf][nr0][16 + hi * 8];
        h8 b10 = *(const h8*)&Bs[buf][nr1][hi * 8];
        h8 b11 = *(const h8*)&Bs[buf][nr1][16 + hi * 8];
        h16 av0 = __builtin_shufflevector(a00, a01, 0,1,2,3,4,5,6,7,8,9,10,11,12,13,14,15);
        h16 av1 = __builtin_shufflevector(a10, a11, 0,1,2,3,4,5,6,7,8,9,10,11,12,13,14,15);
        h16 bv0 = __builtin_shufflevector(b00, b01, 0,1,2,3,4,5,6,7,8,9,10,11,12,13,14,15);
        h16 bv1 = __builtin_shufflevector(b10, b11, 0,1,2,3,4,5,6,7,8,9,10,11,12,13,14,15);

        acc00 = __builtin_amdgcn_wmma_f32_16x16x32_f16(false, av0, false, bv0, (short)0, acc00, false, false);
        acc01 = __builtin_amdgcn_wmma_f32_16x16x32_f16(false, av0, false, bv1, (short)0, acc01, false, false);
        acc10 = __builtin_amdgcn_wmma_f32_16x16x32_f16(false, av1, false, bv0, (short)0, acc10, false, false);
        acc11 = __builtin_amdgcn_wmma_f32_16x16x32_f16(false, av1, false, bv1, (short)0, acc11, false, false);
    }

    // D layout: lane column n = lane&15; VGPR r -> row r + hi*8
    #pragma unroll
    for (int sm = 0; sm < 2; ++sm) {
        #pragma unroll
        for (int sn = 0; sn < 2; ++sn) {
            const v8f acc = (sm == 0) ? (sn == 0 ? acc00 : acc01)
                                      : (sn == 0 ? acc10 : acc11);
            const int opos = posBase + ni2 * 32 + sn * 16 + l16;
            if (opos < Npos) {
                const int mloc = mi2 * 32 + sm * 16 + hi * 8;
                #pragma unroll
                for (int r = 0; r < 8; ++r) {
                    const int m = mBase + mloc + r;
                    if (m < CPout) {
                        float v = acc[r] + biasP[m];
                        if (doRelu) v = fmaxf(v, 0.f);
                        out[(size_t)opos * CPout + m] = (_Float16)v;
                    }
                }
            }
        }
    }
}

// ===========================================================================
// Weight packing: [cout][cin][k][k] f32 -> [Mpad][k*k*CPin] f16 (zero padded)
// ===========================================================================
__global__ void pack_w_kernel(const float* __restrict__ w,
                              const float* __restrict__ bias,
                              _Float16* __restrict__ wpk, float* __restrict__ bpk,
                              int cout, int cin, int ksz, int CPin, int Mpad)
{
    const int Ktot = ksz * ksz * CPin;
    const long long total = (long long)Mpad * Ktot;
    const long long i = (long long)blockIdx.x * blockDim.x + threadIdx.x;
    if (i >= total) return;
    const int m = (int)(i / Ktot);
    const int k = (int)(i - (long long)m * Ktot);
    const int tap = k / CPin;
    const int c   = k - tap * CPin;
    float v = 0.f;
    if (m < cout && c < cin) {
        const int ky = tap / ksz, kx = tap - ky * ksz;
        v = w[(((size_t)m * cin + c) * ksz + ky) * ksz + kx];
    }
    wpk[i] = (_Float16)v;
    if (k == 0) bpk[m] = (m < cout) ? bias[m] : 0.f;
}

// x [B][128][H][W] f32 -> XH [B][H][W][128] f16
__global__ void pack_x_kernel(const float* __restrict__ x, _Float16* __restrict__ xh)
{
    const int i = blockIdx.x * blockDim.x + threadIdx.x;
    if (i >= BB * HWN * DD) return;
    const int c = i & 127;
    const int p = i >> 7;
    const int b = p / HWN;
    const int hw = p - b * HWN;
    xh[i] = (_Float16)x[((size_t)b * DD + c) * HWN + hw];
}

// Build rep tensor [640][384][160] f16: [feat(128), lfg, (a), ind, 0-pad]
__global__ void build_rep_kernel(const _Float16* __restrict__ xh,
                                 const float* __restrict__ lfg,
                                 const float* __restrict__ aprev,
                                 const float* __restrict__ ind,
                                 _Float16* __restrict__ rep, int cinTot)
{
    const long long total = (long long)BKN * HWN * 160;
    const long long i = (long long)blockIdx.x * blockDim.x + threadIdx.x;
    if (i >= total) return;
    const int c = (int)(i % 160);
    const long long p = i / 160;                 // bk*384 + hw
    const int bk = (int)(p / HWN);
    const int hw = (int)(p - (long long)bk * HWN);
    const int b  = bk / KK;
    float v = 0.f;
    if (c < 128)                           v = (float)xh[((size_t)b * HWN + hw) * DD + c];
    else if (c == 128)                     v = lfg[b * HWN + hw];
    else if (aprev != nullptr && c == 129) v = aprev[p];
    else if (c == cinTot - 1)              v = ind[p];
    rep[i] = (_Float16)v;
}

__device__ inline float wave_sum(float v) {
    for (int off = 16; off; off >>= 1) v += __shfl_xor(v, off, 32);
    return v;
}

// ===========================================================================
// Fused lanet tail: out = last_w*( relu(conv5x5_c3(t2)+b3) + conv1x1_r(x0)+rb ) + last_b
// One wave per spatial position (16x24 geometry, pad 2).
// ===========================================================================
__global__ __launch_bounds__(256)
void lanet_tail_kernel(const _Float16* __restrict__ t2, const _Float16* __restrict__ x0,
                       const float* __restrict__ w3, const float* __restrict__ b3,
                       const float* __restrict__ wr, const float* __restrict__ br,
                       const float* __restrict__ wl, const float* __restrict__ bl,
                       float* __restrict__ outp, int Nb, int cin, int CP)
{
    __shared__ float w3s[3328];
    __shared__ float wrs[160];
    const int tid = threadIdx.x;
    for (int i = tid; i < cin * 25; i += 256) w3s[i] = w3[i];
    for (int i = tid; i < cin; i += 256)      wrs[i] = wr[i];
    __syncthreads();

    const int lane = tid & 31, wv = tid >> 5;
    const int pos = blockIdx.x * 8 + wv;
    const int Npos = Nb * HWN;
    float s3 = 0.f, sr = 0.f;
    if (pos < Npos) {
        const int nb = pos / HWN;
        const int r  = pos - nb * HWN;
        const int oy = r / WW, ox = r - (r / WW) * WW;
        for (int tap = 0; tap < 25; ++tap) {
            const int ky = tap / 5, kx = tap - ky * 5;
            const int iy = oy - 2 + ky, ix = ox - 2 + kx;
            if (iy < 0 || iy >= HH || ix < 0 || ix >= WW) continue;
            const _Float16* src = t2 + (size_t)((nb * HH + iy) * WW + ix) * CP;
            for (int c = lane; c < cin; c += 32) s3 += (float)src[c] * w3s[c * 25 + tap];
        }
        const _Float16* xs = x0 + (size_t)pos * CP;
        for (int c = lane; c < cin; c += 32) sr += (float)xs[c] * wrs[c];
    }
    s3 = wave_sum(s3); sr = wave_sum(sr);
    if (pos < Npos && lane == 0) {
        const float o3 = fmaxf(s3 + b3[0], 0.f);
        outp[pos] = wl[0] * (o3 + sr + br[0]) + bl[0];
    }
}

// ===========================================================================
// local-max + top-K + gaussian log-prob maps + log-sigmoid, one block per b
// ===========================================================================
__global__ __launch_bounds__(512)
void localmax_kernel(const float* __restrict__ fgm, float* __restrict__ lfg,
                     float* __restrict__ ind)
{
    __shared__ float satt[HWN];
    __shared__ float slm[HWN];
    __shared__ float rv[512];
    __shared__ int   ri[512];
    __shared__ float sM;
    __shared__ int   sIdx;
    const int t = threadIdx.x;
    const int b = blockIdx.x;
    if (t < HWN) {
        const float v = fgm[b * HWN + t];
        satt[t] = v;
        lfg[b * HWN + t] = fminf(v, 0.f) - log1pf(expf(-fabsf(v)));
    }
    __syncthreads();
    if (t < HWN) {
        const int y = t / WW, x = t - y * WW;
        float mp = -INFINITY;
        for (int dy = -1; dy <= 1; ++dy)
            for (int dx = -1; dx <= 1; ++dx) {
                const int yy = y + dy, xx = x + dx;
                if (yy >= 0 && yy < HH && xx >= 0 && xx < WW)
                    mp = fmaxf(mp, satt[yy * WW + xx]);
            }
        slm[t] = (satt[t] == mp) ? satt[t] : 0.f;   // att * (att==mp)
    }
    __syncthreads();
    for (int k = 0; k < KK; ++k) {
        // argmax (stable: lowest index on tie, matching jax top_k)
        rv[t] = (t < HWN) ? slm[t] : -INFINITY;
        ri[t] = t;
        __syncthreads();
        for (int s = 256; s > 0; s >>= 1) {
            if (t < s) {
                const float v2 = rv[t + s]; const int i2 = ri[t + s];
                if (v2 > rv[t] || (v2 == rv[t] && i2 < ri[t])) { rv[t] = v2; ri[t] = i2; }
            }
            __syncthreads();
        }
        if (t == 0) { sIdx = ri[0]; slm[sIdx] = -INFINITY; }
        __syncthreads();
        const int idx = sIdx;
        const float xp = (float)(idx / WW);
        const float yp = (float)(idx - (idx / WW) * WW);
        float f = 0.f;
        if (t < HWN) {
            const float dh = xp - (float)(t / WW);
            const float dw = yp - (float)(t - (t / WW) * WW);
            f = -0.5f * dh * dh - 0.5f * dw * dw;     // sigma = 1.0
        }
        rv[t] = (t < HWN) ? f : -INFINITY;
        __syncthreads();
        for (int s = 256; s > 0; s >>= 1) { if (t < s) rv[t] = fmaxf(rv[t], rv[t + s]); __syncthreads(); }
        if (t == 0) sM = rv[0];
        __syncthreads();
        const float M = sM;
        rv[t] = (t < HWN) ? expf(f - M) : 0.f;
        __syncthreads();
        for (int s = 256; s > 0; s >>= 1) { if (t < s) rv[t] += rv[t + s]; __syncthreads(); }
        const float lse = M + logf(rv[0]);
        if (t < HWN) ind[((size_t)(b * KK + k)) * HWN + t] = f - lse;
        __syncthreads();
    }
}

// clf c3 (3x3 on 4x4 -> 1x1) + sigmoid. One wave per (b,k).
__global__ void clf_tail_kernel(const _Float16* __restrict__ u2,
                                const float* __restrict__ w, const float* __restrict__ bias,
                                float* __restrict__ wts, float* __restrict__ outw)
{
    const int lane = threadIdx.x & 31, wv = threadIdx.x >> 5;
    const int bk = blockIdx.x * 2 + wv;
    float s = 0.f;
    if (bk < BKN) {
        for (int tap = 0; tap < 9; ++tap) {
            const int ky = tap / 3, kx = tap - ky * 3;
            const _Float16* src = u2 + (size_t)((bk * 4 + ky) * 4 + kx) * 160;
            for (int c = lane; c < 131; c += 32) s += (float)src[c] * w[c * 9 + tap];
        }
    }
    s = wave_sum(s);
    if (bk < BKN && lane == 0) {
        const float v = s + bias[0];
        const float wt = 1.f / (1.f + expf(-v));
        wts[bk] = wt; outw[bk] = wt;
    }
}

__global__ __launch_bounds__(128)
void softmax_kernel(const float* __restrict__ a, float* __restrict__ att)
{
    __shared__ float red[128]; __shared__ float sM, sS;
    const int t = threadIdx.x, bk = blockIdx.x;
    const float* row = a + (size_t)bk * HWN;
    float m = -INFINITY;
    for (int p = t; p < HWN; p += 128) m = fmaxf(m, row[p]);
    red[t] = m; __syncthreads();
    for (int s = 64; s; s >>= 1) { if (t < s) red[t] = fmaxf(red[t], red[t + s]); __syncthreads(); }
    if (t == 0) sM = red[0];
    __syncthreads();
    const float M = sM;
    float ss = 0.f;
    for (int p = t; p < HWN; p += 128) ss += expf(row[p] - M);
    red[t] = ss; __syncthreads();
    for (int s = 64; s; s >>= 1) { if (t < s) red[t] += red[t + s]; __syncthreads(); }
    if (t == 0) sS = red[0];
    __syncthreads();
    const float inv = 1.f / sS;
    for (int p = t; p < HWN; p += 128) att[(size_t)bk * HWN + p] = expf(row[p] - M) * inv;
}

// obj[bk][d] = sum_p att[bk][p]*x[b][d][p]; then l2-normalize over d
__global__ __launch_bounds__(128)
void obj_kernel(const float* __restrict__ att, const float* __restrict__ x,
                float* __restrict__ objn)
{
    __shared__ float sa[HWN]; __shared__ float red[128]; __shared__ float sN;
    const int t = threadIdx.x, bk = blockIdx.x, b = bk / KK;
    for (int p = t; p < HWN; p += 128) sa[p] = att[(size_t)bk * HWN + p];
    __syncthreads();
    const float* xr = x + ((size_t)b * DD + t) * HWN;
    float s = 0.f;
    for (int p = 0; p < HWN; ++p) s += sa[p] * xr[p];
    red[t] = s * s; __syncthreads();
    for (int q = 64; q; q >>= 1) { if (t < q) red[t] += red[t + q]; __syncthreads(); }
    if (t == 0) sN = fmaxf(sqrtf(red[0]), 1e-12f);
    __syncthreads();
    objn[(size_t)bk * DD + t] = s / sN;
}

// o1 = obj@w1^T+b1 ; o2 = obj@w2^T+b2
__global__ __launch_bounds__(128)
void lin_kernel(const float* __restrict__ objn,
                const float* __restrict__ w1, const float* __restrict__ b1,
                const float* __restrict__ w2, const float* __restrict__ b2,
                float* __restrict__ o1, float* __restrict__ o2)
{
    __shared__ float so[DD];
    const int t = threadIdx.x, bk = blockIdx.x;
    so[t] = objn[(size_t)bk * DD + t];
    __syncthreads();
    float s1 = b1[t], s2 = b2[t];
    for (int e = 0; e < DD; ++e) {
        const float o = so[e];
        s1 += o * w1[t * DD + e];
        s2 += o * w2[t * DD + e];
    }
    o1[(size_t)bk * DD + t] = s1;
    o2[(size_t)bk * DD + t] = s2;
}

// pair[b,i,j,:] = where(w[b,i]<0.8, 1, l2norm(o1[b,i]+o2[b,j]))
__global__ __launch_bounds__(128)
void pair_kernel(const float* __restrict__ o1, const float* __restrict__ o2,
                 const float* __restrict__ wts, float* __restrict__ outp)
{
    __shared__ float red[128]; __shared__ float sN;
    const int t = threadIdx.x;
    const int bij = blockIdx.x;
    const int b = bij / 100;
    const int r = bij - b * 100;
    const int i = r / 10, j = r - i * 10;
    const float s = o1[((size_t)b * KK + i) * DD + t] + o2[((size_t)b * KK + j) * DD + t];
    red[t] = s * s; __syncthreads();
    for (int q = 64; q; q >>= 1) { if (t < q) red[t] += red[t + q]; __syncthreads(); }
    if (t == 0) sN = fmaxf(sqrtf(red[0]), 1e-12f);
    __syncthreads();
    const float pv = s / sN;
    const float v = (wts[b * KK + i] < 0.8f) ? 1.0f : pv;
    outp[((size_t)(b * KK + i) * KK + j) * DD + t] = v;
}

__global__ void objout_kernel(const float* __restrict__ objn, const float* __restrict__ wts,
                              float* __restrict__ outp)
{
    const int i = blockIdx.x * blockDim.x + threadIdx.x;
    if (i >= BKN * DD) return;
    outp[i] = (wts[i >> 7] < 0.8f) ? 1.0f : objn[i];
}

// ===========================================================================
// Host launcher
// ===========================================================================
struct LanetP { const float *rw,*rb,*c1w,*c1b,*c2w,*c2b,*c3w,*c3b,*lw,*lb; };

extern "C" void kernel_launch(void* const* d_in, const int* in_sizes, int n_in,
                              void* d_out, int out_size, void* d_ws, size_t ws_size,
                              hipStream_t stream)
{
    (void)n_in; (void)out_size; (void)ws_size;
    const float* x = (const float*)d_in[0];       // [64,128,16,24]

    // ---- parameter mapping (handles insertion-order or sorted pytree flatten)
    LanetP fg, a1, a2, a3;
    const float *clf_c1w,*clf_c1b,*clf_c2w,*clf_c2b,*clf_c3w,*clf_c3b;
    const float *w1,*b1,*w2,*b2;
    auto F = [&](int i){ return (const float*)d_in[i]; };
    const bool sortedLayout = (in_sizes[3] == 130);   // sorted: att1.c1.b first
    if (!sortedLayout) {
        int i = 3;
        auto rd = [&](LanetP& L){ L.rw=F(i++); L.rb=F(i++); L.c1w=F(i++); L.c1b=F(i++);
                                  L.c2w=F(i++); L.c2b=F(i++); L.c3w=F(i++); L.c3b=F(i++);
                                  L.lw=F(i++); L.lb=F(i++); };
        rd(fg); rd(a1); rd(a2); rd(a3);
        clf_c1w=F(i++); clf_c1b=F(i++); clf_c2w=F(i++); clf_c2b=F(i++);
        clf_c3w=F(i++); clf_c3b=F(i++);
        w1=F(i++); b1=F(i++); w2=F(i++); b2=F(i++);
    } else {
        int i = 3;
        auto rd = [&](LanetP& L){ L.c1b=F(i++); L.c1w=F(i++); L.c2b=F(i++); L.c2w=F(i++);
                                  L.c3b=F(i++); L.c3w=F(i++); L.lb=F(i++); L.lw=F(i++);
                                  L.rb=F(i++); L.rw=F(i++); };
        rd(a1); rd(a2); rd(a3);
        b1=F(i++); b2=F(i++);
        clf_c1b=F(i++); clf_c1w=F(i++); clf_c2b=F(i++); clf_c2w=F(i++);
        clf_c3b=F(i++); clf_c3w=F(i++);
        rd(fg);
        w1=F(i++); w2=F(i++);
    }

    // ---- workspace layout
    char* wsb = (char*)d_ws;
    size_t off = 0;
    auto alloc = [&](size_t bytes) -> void* {
        void* p = wsb + off;
        off = (off + bytes + 255) & ~(size_t)255;
        return p;
    };
    _Float16* XH   = (_Float16*)alloc((size_t)BB * HWN * DD * 2);
    _Float16* BIG0 = (_Float16*)alloc((size_t)BKN * HWN * 160 * 2);   // rep
    _Float16* BIG1 = (_Float16*)alloc((size_t)BKN * HWN * 160 * 2);   // t1 / clf u1
    _Float16* BIG2 = (_Float16*)alloc((size_t)BKN * HWN * 160 * 2);   // t2 / clf u2
    float* fgm  = (float*)alloc((size_t)BB * HWN * 4);
    float* lfg  = (float*)alloc((size_t)BB * HWN * 4);
    float* ind  = (float*)alloc((size_t)BKN * HWN * 4);
    float* abuf = (float*)alloc((size_t)BKN * HWN * 4);
    float* attb = (float*)alloc((size_t)BKN * HWN * 4);
    float* objn = (float*)alloc((size_t)BKN * DD * 4);
    float* o1   = (float*)alloc((size_t)BKN * DD * 4);
    float* o2   = (float*)alloc((size_t)BKN * DD * 4);
    float* wts  = (float*)alloc((size_t)BKN * 4);
    _Float16* PKW = (_Float16*)alloc((size_t)192 * 4000 * 2);
    float*    PKB = (float*)alloc(192 * 4);

    float* out = (float*)d_out;
    float* out_obj  = out;             // 64*10*128
    float* out_pair = out + 81920;     // 64*10*10*128
    float* out_w    = out + 901120;    // 640

    auto packW = [&](const float* w, const float* b, int cout, int cin,
                     int ksz, int CPin, int Mpad) {
        const long long tot = (long long)Mpad * ksz * ksz * CPin;
        pack_w_kernel<<<dim3((unsigned)((tot + 255) / 256)), dim3(256), 0, stream>>>(
            w, b, PKW, PKB, cout, cin, ksz, CPin, Mpad);
    };
    auto conv = [&](const _Float16* in_, _Float16* out_, int Nb, int Hin, int Win,
                    int Hout, int Wout, int CPin, int CPout, int ksz, int strd,
                    int padH, int padW, int Mpad, int relu) {
        const int Npos = Nb * Hout * Wout;
        dim3 g((Npos + 127) / 128, Mpad / 64);
        conv_wmma_kernel<<<g, 256, 0, stream>>>(in_, PKW, PKB, out_,
            Nb, Hin, Win, Hout, Wout, CPin, CPout, ksz, strd, padH, padW, relu);
    };
    auto tail = [&](const _Float16* t2, const _Float16* x0, const LanetP& L,
                    float* outp, int Nb, int cin, int CP) {
        const int Npos = Nb * HWN;
        lanet_tail_kernel<<<dim3((Npos + 7) / 8), 256, 0, stream>>>(
            t2, x0, L.c3w, L.c3b, L.rw, L.rb, L.lw, L.lb, outp, Nb, cin, CP);
    };
    const long long repTot = (long long)BKN * HWN * 160;
    auto buildRep = [&](const float* aprev, int cinTot) {
        build_rep_kernel<<<dim3((unsigned)((repTot + 255) / 256)), 256, 0, stream>>>(
            XH, lfg, aprev, ind, BIG0, cinTot);
    };

    // ---- pipeline --------------------------------------------------------
    pack_x_kernel<<<dim3((BB * HWN * DD + 255) / 256), 256, 0, stream>>>(x, XH);

    // fg lanet (cin=128)
    packW(fg.c1w, fg.c1b, 128, 128, 5, 128, 128);
    conv(XH, BIG1, BB, HH, WW, HH, WW, 128, 128, 5, 1, 2, 2, 128, 1);
    packW(fg.c2w, fg.c2b, 128, 128, 5, 128, 128);
    conv(BIG1, BIG2, BB, HH, WW, HH, WW, 128, 128, 5, 1, 2, 2, 128, 1);
    tail(BIG2, XH, fg, fgm, BB, 128, 128);

    localmax_kernel<<<dim3(BB), 512, 0, stream>>>(fgm, lfg, ind);

    // att1 (cin=130)
    buildRep(nullptr, 130);
    packW(a1.c1w, a1.c1b, 130, 130, 5, 160, 192);
    conv(BIG0, BIG1, BKN, HH, WW, HH, WW, 160, 160, 5, 1, 2, 2, 192, 1);
    packW(a1.c2w, a1.c2b, 130, 130, 5, 160, 192);
    conv(BIG1, BIG2, BKN, HH, WW, HH, WW, 160, 160, 5, 1, 2, 2, 192, 1);
    tail(BIG2, BIG0, a1, abuf, BKN, 130, 160);

    // att2 (cin=131)
    buildRep(abuf, 131);
    packW(a2.c1w, a2.c1b, 131, 131, 5, 160, 192);
    conv(BIG0, BIG1, BKN, HH, WW, HH, WW, 160, 160, 5, 1, 2, 2, 192, 1);
    packW(a2.c2w, a2.c2b, 131, 131, 5, 160, 192);
    conv(BIG1, BIG2, BKN, HH, WW, HH, WW, 160, 160, 5, 1, 2, 2, 192, 1);
    tail(BIG2, BIG0, a2, abuf, BKN, 131, 160);

    // att3 (cin=131) -- BIG0 keeps rep3 for the classifier
    buildRep(abuf, 131);
    packW(a3.c1w, a3.c1b, 131, 131, 5, 160, 192);
    conv(BIG0, BIG1, BKN, HH, WW, HH, WW, 160, 160, 5, 1, 2, 2, 192, 1);
    packW(a3.c2w, a3.c2b, 131, 131, 5, 160, 192);
    conv(BIG1, BIG2, BKN, HH, WW, HH, WW, 160, 160, 5, 1, 2, 2, 192, 1);
    tail(BIG2, BIG0, a3, abuf, BKN, 131, 160);    // abuf = a3

    // classifier on rep3: 16x24 -> 10x10 -> 4x4 -> 1x1
    packW(clf_c1w, clf_c1b, 131, 131, 5, 160, 192);
    conv(BIG0, BIG1, BKN, HH, WW, 10, 10, 160, 160, 5, 2, 4, 0, 192, 1);
    packW(clf_c2w, clf_c2b, 131, 131, 3, 160, 192);
    conv(BIG1, BIG2, BKN, 10, 10, 4, 4, 160, 160, 3, 2, 0, 0, 192, 1);
    clf_tail_kernel<<<dim3(BKN / 2), 64, 0, stream>>>(BIG2, clf_c3w, clf_c3b, wts, out_w);

    // softmax(a3), obj, linears, pair, outputs
    softmax_kernel<<<dim3(BKN), 128, 0, stream>>>(abuf, attb);
    obj_kernel<<<dim3(BKN), 128, 0, stream>>>(attb, x, objn);
    lin_kernel<<<dim3(BKN), 128, 0, stream>>>(objn, w1, b1, w2, b2, o1, o2);
    objout_kernel<<<dim3((BKN * DD + 255) / 256), 256, 0, stream>>>(objn, wts, out_obj);
    pair_kernel<<<dim3(BB * KK * KK), 128, 0, stream>>>(o1, o2, wts, out_pair);
}
